// TextureCSEFixed_34110630265641
// MI455X (gfx1250) — matrix operand order repaired and checked by reading, not compiled, not run on hardware
//
#include <hip/hip_runtime.h>

typedef __attribute__((ext_vector_type(16))) _Float16 v16h;
typedef __attribute__((ext_vector_type(8)))  _Float16 v8h;
typedef __attribute__((ext_vector_type(8)))  float    v8f;

#if defined(__has_builtin)
#if __has_builtin(__builtin_amdgcn_sched_barrier)
#define SCHED_FENCE() __builtin_amdgcn_sched_barrier(0x14)  // SALU|VMEM may cross
#endif
#endif
#ifndef SCHED_FENCE
#define SCHED_FENCE()
#endif

// ---------------------------------------------------------------------------
// Prep: build the padded B stream over Npad = roundup(N, 64) vertices.
//   Yh  row n: Y[n] as f16 (zeros for pad rows)
//   Ysq row n: [bias, 0, ..., 0]; bias = 0.5*||Y[n]||^2 for real rows,
//              65504 (f16 max) for pad rows so they can never win the argmin.
// Ysq rows feed lanes 16..31 (K=16..31): K=16 carries the bias, picked up by
// A's constant 1.0 at K=16, so the WMMA accumulator input is inline 0.
// ---------------------------------------------------------------------------
__global__ void prep_verts_kernel(const float* __restrict__ Y,
                                  _Float16* __restrict__ Yh,
                                  _Float16* __restrict__ Ysq,
                                  int N, int Npad) {
  int n = blockIdx.x * blockDim.x + threadIdx.x;
  if (n >= Npad) return;
  v8h lo = {}, hi = {};
  float bias = 65504.0f;  // pad rows: huge score, never selected
  if (n < N) {
    const float* yp = Y + (size_t)n * 16;
    float s = 0.f;
#pragma unroll
    for (int k = 0; k < 8; ++k) {
      float v0 = yp[k], v1 = yp[k + 8];
      s += v0 * v0 + v1 * v1;
      lo[k] = (_Float16)v0;
      hi[k] = (_Float16)v1;
    }
    bias = 0.5f * s;
  }
  *(v8h*)(Yh + (size_t)n * 16)     = lo;
  *(v8h*)(Yh + (size_t)n * 16 + 8) = hi;
  v8h q = {};
  q[0] = (_Float16)bias;
  v8h z = {};
  *(v8h*)(Ysq + (size_t)n * 16)     = q;
  *(v8h*)(Ysq + (size_t)n * 16 + 8) = z;
}

// ---------------------------------------------------------------------------
__device__ __forceinline__ v8f wmma_tile(v16h a, const _Float16* p) {
  v16h b = *(const v16h*)p;
  v8f cz = {};
  return __builtin_amdgcn_wmma_f32_16x16x32_f16(false, a, false, b,
                                                (short)0, cz, false, false);
}

// Tag scores with the column index (low 12 mantissa bits, Npad <= 4096) and
// keep a running packed min. The empty asm makes `n` an opaque VGPR so the
// compiler emits one fused v_and_or_b32 per element; pairs of fminf fuse to
// v_min3_num_f32.
__device__ __forceinline__ void consume_tile(const v8f& c, int n, float best[8]) {
  asm("" : "+v"(n));
#pragma unroll
  for (int r = 0; r < 8; ++r) {
    float packed = __int_as_float((__float_as_int(c[r]) & 0xFFFFF000) | n);
    best[r] = fminf(best[r], packed);
  }
}

// ---------------------------------------------------------------------------
// Main: one wave owns 16 rows of X. A = (-X | 1.0 at K=16) held in VGPRs for
// all column tiles. Per tile: D = (-X)*Y^T + 0.5*||y||^2 in one WMMA with
// inline-0 C. Modulo schedule pinned by sched_barriers (VALU/WMMA cannot
// cross, VMEM can): [consume c0,c1][wmma c0,c1][consume c2,c3][wmma c2,c3].
// Each consume reads a WMMA two groups back (hazard covered, no v_nops), and
// each accumulator dies right before its redef (no rotation copies). nTiles
// is a multiple of 4 by construction (padded B stream).
// ---------------------------------------------------------------------------
__global__ void __launch_bounds__(256)
nn_argmin_kernel(const float* __restrict__ X,
                 const float* __restrict__ colors,
                 const _Float16* __restrict__ Bfrag,  // [Yh | Ysq], each Npad*16
                 float* __restrict__ out,
                 int rowTiles, int nTiles, int Npad) {
  const int lane = threadIdx.x & 31;
  const int wave = threadIdx.x >> 5;
  const int rowTile = blockIdx.x * 8 + wave;
  if (rowTile >= rowTiles) return;
  const int rowBase = rowTile * 16;
  const int col  = lane & 15;   // A: row within tile; B/C: column within tile
  const int half = lane >> 4;   // K-half for A/B, row-half for C

  // A fragment (contiguous K order): element e holds K = 16*half + e.
  // half==0 lanes hold -X (K=0..15); half==1 lanes set K=16 to 1.0 so the
  // WMMA adds 1.0 * bias from B's K=16 slot.
  v16h a = {};
  {
    const float* xp = X + (size_t)(rowBase + col) * 16;
#pragma unroll
    for (int k = 0; k < 16; ++k) a[k] = half == 0 ? (_Float16)(-xp[k]) : (_Float16)0.0f;
    if (half == 1) a[0] = (_Float16)1.0f;
  }

  // B source: lanes 0..15 stream Yh rows (K=0..15), lanes 16..31 stream Ysq
  // rows (K=16..31). Uniform 256-half stride per column tile.
  const _Float16* bp =
      Bfrag + (size_t)half * (size_t)Npad * 16 + (size_t)col * 16;

  float best[8];
#pragma unroll
  for (int r = 0; r < 8; ++r) best[r] = __int_as_float(0x7f000000);

  // Prologue: 4 tiles in flight (nTiles >= 4 guaranteed by padding).
  v8f c0 = wmma_tile(a, bp);
  v8f c1 = wmma_tile(a, bp + 256);
  v8f c2 = wmma_tile(a, bp + 512);
  v8f c3 = wmma_tile(a, bp + 768);
  int t = 4;
  for (; t + 3 < nTiles; t += 4) {
    consume_tile(c0, (t - 4) * 16 + col, best);
    consume_tile(c1, (t - 3) * 16 + col, best);
    SCHED_FENCE();
    c0 = wmma_tile(a, bp + (size_t)t * 256);
    c1 = wmma_tile(a, bp + (size_t)(t + 1) * 256);
    SCHED_FENCE();
    consume_tile(c2, (t - 2) * 16 + col, best);
    consume_tile(c3, (t - 1) * 16 + col, best);
    SCHED_FENCE();
    c2 = wmma_tile(a, bp + (size_t)(t + 2) * 256);
    c3 = wmma_tile(a, bp + (size_t)(t + 3) * 256);
    SCHED_FENCE();
  }
  consume_tile(c0, (t - 4) * 16 + col, best);
  consume_tile(c1, (t - 3) * 16 + col, best);
  consume_tile(c2, (t - 2) * 16 + col, best);
  consume_tile(c3, (t - 1) * 16 + col, best);

  // Reduce across the 16 columns of each half-wave (masks <16 stay in-half);
  // exact on the packed (score,index) keys.
#pragma unroll
  for (int r = 0; r < 8; ++r) {
#pragma unroll
    for (int m = 8; m >= 1; m >>= 1)
      best[r] = fminf(best[r], __shfl_xor(best[r], m, 32));
  }

  // Lane 0 holds rows rowBase+0..7, lane 16 holds rows rowBase+8..15.
  if (col == 0) {
    const int base = rowBase + half * 8;
#pragma unroll
    for (int r = 0; r < 8; ++r) {
      const int v = __float_as_int(best[r]) & 0xFFF;
      out[(size_t)(base + r) * 3 + 0] = colors[(size_t)v * 3 + 0];
      out[(size_t)(base + r) * 3 + 1] = colors[(size_t)v * 3 + 1];
      out[(size_t)(base + r) * 3 + 2] = colors[(size_t)v * 3 + 2];
    }
  }
}

// ---------------------------------------------------------------------------
extern "C" void kernel_launch(void* const* d_in, const int* in_sizes, int n_in,
                              void* d_out, int out_size, void* d_ws, size_t ws_size,
                              hipStream_t stream) {
  const float* X      = (const float*)d_in[0];  // (M,16) f32
  const float* colors = (const float*)d_in[1];  // (N,3)  f32
  const float* Y      = (const float*)d_in[2];  // (N,16) f32
  const int M = in_sizes[0] / 16;
  const int N = in_sizes[2] / 16;
  const int Npad = (N + 63) & ~63;  // tile count multiple of 4 (4000 -> 4032)

  // Workspace layout (32B aligned): Yh (Npad*32 B) | Ysq (Npad*32 B)
  char* ws = (char*)d_ws;
  _Float16* Yh  = (_Float16*)ws;
  _Float16* Ysq = (_Float16*)(ws + (size_t)Npad * 32);

  prep_verts_kernel<<<(Npad + 255) / 256, 256, 0, stream>>>(Y, Yh, Ysq, N, Npad);

  const int rowTiles = M / 16;             // M = 131072 -> 8192 waves
  const int blocks   = (rowTiles + 7) / 8; // 8 waves (256 thr) per block
  nn_argmin_kernel<<<blocks, 256, 0, stream>>>(X, colors, Yh, (float*)d_out,
                                               rowTiles, Npad / 16, Npad);
}
